// CovetCelltype_7842610282626
// MI455X (gfx1250) — matrix operand order repaired and checked by reading, not compiled, not run on hardware
//
#include <hip/hip_runtime.h>

// ---------- types ----------
typedef __attribute__((ext_vector_type(16))) __bf16 v16bf;
typedef __attribute__((ext_vector_type(8)))  __bf16 v8bf;
typedef __attribute__((ext_vector_type(8)))  float  v8f;

#define N_CELLS 16384
#define G       32
#define KNN     8
#define IN_DIM  2000
#define HID_K   3024   // real hidden features (2000 + 1024)
#define HID_P   3072   // padded K for bf16 WMMA (multiple of 32)
#define H1      1024
#define OUT_D   30
#define OUT_P   32

static __device__ __forceinline__ __bf16 f2bf(float f) {
  unsigned u = __builtin_bit_cast(unsigned, f);
  unsigned r = u + 0x7FFFu + ((u >> 16) & 1u);   // round-to-nearest-even
  unsigned short h = (unsigned short)(r >> 16);
  return __builtin_bit_cast(__bf16, h);
}

// ---- CDNA5 async global->LDS copies (ASYNCcnt path), GVS addressing ----
static __device__ __forceinline__ void async_b128(unsigned lds_off, const void* base,
                                                  unsigned byte_off) {
  asm volatile("global_load_async_to_lds_b128 %0, %1, %2"
               :: "v"(lds_off), "v"(byte_off), "s"(base) : "memory");
}
static __device__ __forceinline__ void async_b64(unsigned lds_off, const void* base,
                                                 unsigned byte_off) {
  asm volatile("global_load_async_to_lds_b64 %0, %1, %2"
               :: "v"(lds_off), "v"(byte_off), "s"(base) : "memory");
}
static __device__ __forceinline__ void wait_async0() {
  asm volatile("s_wait_asynccnt 0x0" ::: "memory");
}
static __device__ __forceinline__ unsigned lds_addr(const void* p) {
  return (unsigned)(uintptr_t)p;   // low 32 bits of flat address == LDS offset
}

// ---------------------------------------------------------------------------
// K1: global mean of genes  [N,32] -> [32]
// ---------------------------------------------------------------------------
__global__ __launch_bounds__(256) void mean_genes(const float* __restrict__ genes,
                                                  float* __restrict__ mean) {
  __shared__ float s[G];
  int t = threadIdx.x;
  if (t < G) s[t] = 0.f;
  __syncthreads();
  float loc[G];
#pragma unroll
  for (int g = 0; g < G; ++g) loc[g] = 0.f;
  for (int r = t; r < N_CELLS; r += 256) {
    const float* row = genes + (size_t)r * G;
#pragma unroll
    for (int g = 0; g < G; ++g) loc[g] += row[g];
  }
#pragma unroll
  for (int g = 0; g < G; ++g) atomicAdd(&s[g], loc[g]);
  __syncthreads();
  if (t < G) mean[t] = s[t] * (1.f / (float)N_CELLS);
}

// ---------------------------------------------------------------------------
// K2: brute-force 8-NN over 2D coords (incl. self), one thread per cell
// ---------------------------------------------------------------------------
__global__ __launch_bounds__(256) void knn_kernel(const float* __restrict__ coords,
                                                  int* __restrict__ nidx) {
  __shared__ float cx[256], cy[256];
  int i = blockIdx.x * 256 + threadIdx.x;
  float xi = coords[2 * i], yi = coords[2 * i + 1];
  float bd[KNN]; int bi[KNN];
#pragma unroll
  for (int q = 0; q < KNN; ++q) { bd[q] = 3.4e38f; bi[q] = 0; }
  for (int tile = 0; tile < N_CELLS; tile += 256) {
    __syncthreads();
    cx[threadIdx.x] = coords[2 * (tile + threadIdx.x)];
    cy[threadIdx.x] = coords[2 * (tile + threadIdx.x) + 1];
    __syncthreads();
    for (int j = 0; j < 256; ++j) {
      float dx = xi - cx[j], dy = yi - cy[j];
      float d2 = dx * dx + dy * dy;
      if (d2 < bd[KNN - 1]) {           // rare path: bubble-insert (all static idx)
        int jj = tile + j;
#pragma unroll
        for (int q = 0; q < KNN; ++q) {
          bool c = d2 < bd[q];
          float td = c ? bd[q] : d2;  int ti = c ? bi[q] : jj;
          bd[q]    = c ? d2    : bd[q];
          bi[q]    = c ? jj    : bi[q];
          d2 = td; jj = ti;
        }
      }
    }
  }
#pragma unroll
  for (int q = 0; q < KNN; ++q) nidx[i * KNN + q] = bi[q];
}

// ---------------------------------------------------------------------------
// K3: COVET sqrt via rank-8 Gram trick, one wave per cell, lane = gene column.
//     sqrt(C^T C)/sqrt(k-1) = B^T B / sqrt(7),  B = D^{-1/4} U^T C,  CC^T=UDU^T
//     Writes bf16 directly into hidden[:, 2000:3024].
// ---------------------------------------------------------------------------
__global__ __launch_bounds__(256) void covet_kernel(const float* __restrict__ genes,
                                                    const int* __restrict__ nidx,
                                                    const float* __restrict__ mean,
                                                    __bf16* __restrict__ hid) {
  __shared__ float Sm[8][64];
  __shared__ float Um[8][64];
  __shared__ float Fv[8][8];
  __shared__ float Bm[8][8 * G];
  int w = threadIdx.x >> 5, lane = threadIdx.x & 31;
  int cell = blockIdx.x * 8 + w;
  float mg = mean[lane];
  float c[KNN];
#pragma unroll
  for (int i = 0; i < KNN; ++i) {
    int row = nidx[cell * KNN + i];
    c[i] = genes[(size_t)row * G + lane] - mg;
  }
  // S = C C^T  (8x8) via wave butterfly reductions
#pragma unroll
  for (int i = 0; i < KNN; ++i) {
#pragma unroll
    for (int j = i; j < KNN; ++j) {
      float v = c[i] * c[j];
      v += __shfl_xor(v, 1, 32);  v += __shfl_xor(v, 2, 32);
      v += __shfl_xor(v, 4, 32);  v += __shfl_xor(v, 8, 32);
      v += __shfl_xor(v, 16, 32);
      if (lane == 0) { Sm[w][i * 8 + j] = v; Sm[w][j * 8 + i] = v; }
    }
  }
  __syncthreads();
  if (lane == 0) {                      // serial 8x8 cyclic Jacobi
    float* S = &Sm[w][0];
    float* U = &Um[w][0];
    for (int i = 0; i < 64; ++i) U[i] = ((i >> 3) == (i & 7)) ? 1.f : 0.f;
    for (int sweep = 0; sweep < 10; ++sweep) {
      for (int p = 0; p < 7; ++p) {
        for (int q = p + 1; q < 8; ++q) {
          float apq = S[p * 8 + q];
          if (fabsf(apq) < 1e-12f) continue;
          float app = S[p * 8 + p], aqq = S[q * 8 + q];
          float theta = 0.5f * (aqq - app) / apq;
          float tt = (theta >= 0.f ? 1.f : -1.f) /
                     (fabsf(theta) + sqrtf(1.f + theta * theta));
          float cth = rsqrtf(1.f + tt * tt);
          float sth = tt * cth;
          for (int k2 = 0; k2 < 8; ++k2) {          // S <- S*G
            float a = S[k2 * 8 + p], b = S[k2 * 8 + q];
            S[k2 * 8 + p] = cth * a - sth * b;
            S[k2 * 8 + q] = sth * a + cth * b;
          }
          for (int k2 = 0; k2 < 8; ++k2) {          // S <- G^T*S
            float a = S[p * 8 + k2], b = S[q * 8 + k2];
            S[p * 8 + k2] = cth * a - sth * b;
            S[q * 8 + k2] = sth * a + cth * b;
          }
          for (int k2 = 0; k2 < 8; ++k2) {          // U <- U*G
            float a = U[k2 * 8 + p], b = U[k2 * 8 + q];
            U[k2 * 8 + p] = cth * a - sth * b;
            U[k2 * 8 + q] = sth * a + cth * b;
          }
        }
      }
    }
    for (int i = 0; i < 8; ++i) {
      float d = S[i * 8 + i];
      Fv[w][i] = (d > 1e-10f) ? rsqrtf(sqrtf(d)) : 0.f;   // d^{-1/4} (clamped)
    }
  }
  __syncthreads();
  float bg[KNN];
#pragma unroll
  for (int i = 0; i < KNN; ++i) {
    float acc = 0.f;
#pragma unroll
    for (int j = 0; j < KNN; ++j) acc += Um[w][j * 8 + i] * c[j];
    bg[i] = acc * Fv[w][i];
    Bm[w][i * G + lane] = bg[i];
  }
  __syncthreads();
  const float rs7 = 0.3779644730092272f;   // 1/sqrt(k-1)
  __bf16* dst = hid + (size_t)cell * HID_P + IN_DIM;
  for (int r = 0; r < G; ++r) {
    float acc = 0.f;
#pragma unroll
    for (int i = 0; i < KNN; ++i) acc += Bm[w][i * G + r] * bg[i];
    dst[r * G + lane] = f2bf(acc * rs7);
  }
}

// ---------------------------------------------------------------------------
// K4: pack x (fp32 -> bf16) into hidden[:, 0:2000]; zero tail pad [3024:3072)
// ---------------------------------------------------------------------------
__global__ __launch_bounds__(256) void pack_hidden(const float* __restrict__ x,
                                                   __bf16* __restrict__ hid) {
  size_t t = (size_t)blockIdx.x * 256 + threadIdx.x;
  size_t n = t / HID_P;
  int cidx = (int)(t % HID_P);
  if (n >= N_CELLS) return;
  if (cidx < IN_DIM)       hid[t] = f2bf(x[n * IN_DIM + cidx]);
  else if (cidx >= HID_K)  hid[t] = f2bf(0.f);
}

// ---------------------------------------------------------------------------
// K5: weights -> bf16, TRANSPOSED (N-major) with K / N zero padding.
//     W1t[n][k] : [1024][3072],  W2t[n][k] : [32][1024]
// ---------------------------------------------------------------------------
__global__ __launch_bounds__(256) void prep_weights(const float* __restrict__ W1,
                                                    const float* __restrict__ W2,
                                                    __bf16* __restrict__ W1t,
                                                    __bf16* __restrict__ W2t) {
  size_t t = (size_t)blockIdx.x * 256 + threadIdx.x;
  const size_t n1 = (size_t)H1 * HID_P;
  if (t < n1) {
    int n = (int)(t / HID_P), k = (int)(t % HID_P);
    W1t[t] = f2bf(k < HID_K ? W1[(size_t)k * H1 + n] : 0.f);
  } else if (t < n1 + (size_t)OUT_P * H1) {
    size_t u = t - n1;
    int n = (int)(u / H1), k = (int)(u % H1);
    W2t[u] = f2bf(n < OUT_D ? W2[(size_t)k * OUT_D + n] : 0.f);
  }
}

// ---------------------------------------------------------------------------
// K6: GEMM1 + bias + ReLU:  h = relu(hidden[N,3072] @ W1 + b1)
//     bf16 WMMA 16x16x32, f32 accumulate. 8 waves, block tile 128x64, 32x32
//     per wave. Double-buffered async global->LDS copies (ASYNCcnt).
// ---------------------------------------------------------------------------
__global__ __launch_bounds__(256) void gemm1_relu(const __bf16* __restrict__ A,
                                                  const __bf16* __restrict__ Bt,
                                                  const float* __restrict__ bias,
                                                  __bf16* __restrict__ Hout) {
  constexpr int AP = 40;                        // LDS pitch: 80B, 16B-aligned
  __shared__ alignas(16) __bf16 Al[2][128 * AP];   // A tiles 128x32
  __shared__ alignas(16) __bf16 Bl[2][64 * AP];    // B tiles (col-major K) 64x32
  int tid = threadIdx.x;
  int w = tid >> 5, lane = tid & 31;
  int wm = w >> 1, wn = w & 1;
  int mbase = blockIdx.y * 128;
  int nbase = blockIdx.x * 64;

  v8f acc[2][2] = {};

  // copy assignments: A 32B/thread (2 x b128), B 16B/thread (1 x b128)
  int arow = tid >> 1, ahalf = (tid & 1) * 16;
  int bc = tid >> 2,  bh = (tid & 3) * 8;
  unsigned a_lds[2], b_lds[2];
  a_lds[0] = lds_addr(&Al[0][arow * AP + ahalf]);
  a_lds[1] = lds_addr(&Al[1][arow * AP + ahalf]);
  b_lds[0] = lds_addr(&Bl[0][bc * AP + bh]);
  b_lds[1] = lds_addr(&Bl[1][bc * AP + bh]);
  unsigned a_goff = (unsigned)(((size_t)(mbase + arow) * HID_P + ahalf) * 2);
  unsigned b_goff = (unsigned)(((size_t)(nbase + bc) * HID_P + bh) * 2);

  // prologue: fill buffer 0
  async_b128(a_lds[0],      A,  a_goff);
  async_b128(a_lds[0] + 16, A,  a_goff + 16);
  async_b128(b_lds[0],      Bt, b_goff);

  int lrow = lane & 15;
  int ko = (lane < 16) ? 0 : 8;          // A-frag K origin per half-wave
  int ks = (lane < 16) ? 0 : 8;          // B-frag K origin per half-wave
  int ai[2], bidx[2];
  ai[0]   = (wm * 32 + lrow) * AP + ko;   ai[1]   = ai[0]   + 16 * AP;
  bidx[0] = (wn * 32 + lrow) * AP + ks;   bidx[1] = bidx[0] + 16 * AP;

  for (int kk = 0; kk < HID_P; kk += 32) {
    int cur = (kk >> 5) & 1;
    wait_async0();                       // my copies into `cur` are done
    __syncthreads();                     // everyone's are; prior reads finished
    if (kk + 32 < HID_P) {               // overlap: fill next buffer
      int nxt = cur ^ 1;
      unsigned kb = (unsigned)(kk + 32) * 2;
      async_b128(a_lds[nxt],      A,  a_goff + kb);
      async_b128(a_lds[nxt] + 16, A,  a_goff + kb + 16);
      async_b128(b_lds[nxt],      Bt, b_goff + kb);
    }
    const __bf16* Ac = Al[cur];
    const __bf16* Bc = Bl[cur];
    v16bf af[2], bfr[2];
#pragma unroll
    for (int mt = 0; mt < 2; ++mt) {
      v8bf lo = *(const v8bf*)(Ac + ai[mt]);
      v8bf hi = *(const v8bf*)(Ac + ai[mt] + 16);
      af[mt] = __builtin_shufflevector(lo, hi, 0,1,2,3,4,5,6,7,8,9,10,11,12,13,14,15);
    }
#pragma unroll
    for (int nt = 0; nt < 2; ++nt) {
      v8bf lo = *(const v8bf*)(Bc + bidx[nt]);
      v8bf hi = *(const v8bf*)(Bc + bidx[nt] + 16);
      bfr[nt] = __builtin_shufflevector(lo, hi, 0,1,2,3,4,5,6,7,8,9,10,11,12,13,14,15);
    }
#pragma unroll
    for (int mt = 0; mt < 2; ++mt)
#pragma unroll
      for (int nt = 0; nt < 2; ++nt)
        acc[mt][nt] = __builtin_amdgcn_wmma_f32_16x16x32_bf16(
            false, af[mt], false, bfr[nt], (short)0, acc[mt][nt], false, false);
  }

  int roff = (lane < 16) ? 0 : 8;
#pragma unroll
  for (int mt = 0; mt < 2; ++mt)
#pragma unroll
    for (int nt = 0; nt < 2; ++nt) {
      int col = nbase + wn * 32 + nt * 16 + lrow;
      float bb = bias[col];
#pragma unroll
      for (int j = 0; j < 8; ++j) {
        int row = mbase + wm * 32 + mt * 16 + roff + j;
        float v = acc[mt][nt][j] + bb;
        v = v > 0.f ? v : 0.f;
        Hout[(size_t)row * H1 + col] = f2bf(v);
      }
    }
}

// ---------------------------------------------------------------------------
// K7: GEMM2 + bias:  out[N,30] = h[N,1024] @ W2 + b2   (fp32 out)
//     Same double-buffered async structure, wave tile 16x32.
// ---------------------------------------------------------------------------
__global__ __launch_bounds__(256) void gemm2_bias(const __bf16* __restrict__ A,
                                                  const __bf16* __restrict__ Bt,
                                                  const float* __restrict__ bias,
                                                  float* __restrict__ out) {
  constexpr int AP = 40;
  __shared__ alignas(16) __bf16 Al[2][128 * AP];   // A tiles 128x32
  __shared__ alignas(16) __bf16 Bl[2][32 * AP];    // B tiles (col-major K) 32x32
  int tid = threadIdx.x;
  int w = tid >> 5, lane = tid & 31;
  int mbase = blockIdx.x * 128;
  v8f acc[2] = {};

  int arow = tid >> 1, ahalf = (tid & 1) * 16;
  int bc = tid >> 3,  bh = (tid & 7) * 4;          // B: 8B/thread (1 x b64)
  unsigned a_lds[2], b_lds[2];
  a_lds[0] = lds_addr(&Al[0][arow * AP + ahalf]);
  a_lds[1] = lds_addr(&Al[1][arow * AP + ahalf]);
  b_lds[0] = lds_addr(&Bl[0][bc * AP + bh]);
  b_lds[1] = lds_addr(&Bl[1][bc * AP + bh]);
  unsigned a_goff = (unsigned)(((size_t)(mbase + arow) * H1 + ahalf) * 2);
  unsigned b_goff = (unsigned)(((size_t)bc * H1 + bh) * 2);

  async_b128(a_lds[0],      A,  a_goff);
  async_b128(a_lds[0] + 16, A,  a_goff + 16);
  async_b64 (b_lds[0],      Bt, b_goff);

  int lrow = lane & 15;
  int ko = (lane < 16) ? 0 : 8;
  int ks = (lane < 16) ? 0 : 8;
  int aix = (w * 16 + lrow) * AP + ko;
  int bidx[2];
  bidx[0] = lrow * AP + ks;  bidx[1] = bidx[0] + 16 * AP;

  for (int kk = 0; kk < H1; kk += 32) {
    int cur = (kk >> 5) & 1;
    wait_async0();
    __syncthreads();
    if (kk + 32 < H1) {
      int nxt = cur ^ 1;
      unsigned kb = (unsigned)(kk + 32) * 2;
      async_b128(a_lds[nxt],      A,  a_goff + kb);
      async_b128(a_lds[nxt] + 16, A,  a_goff + kb + 16);
      async_b64 (b_lds[nxt],      Bt, b_goff + kb);
    }
    const __bf16* Ac = Al[cur];
    const __bf16* Bc = Bl[cur];
    v16bf af;
    {
      v8bf lo = *(const v8bf*)(Ac + aix);
      v8bf hi = *(const v8bf*)(Ac + aix + 16);
      af = __builtin_shufflevector(lo, hi, 0,1,2,3,4,5,6,7,8,9,10,11,12,13,14,15);
    }
#pragma unroll
    for (int nt = 0; nt < 2; ++nt) {
      v8bf lo = *(const v8bf*)(Bc + bidx[nt]);
      v8bf hi = *(const v8bf*)(Bc + bidx[nt] + 16);
      v16bf bfr = __builtin_shufflevector(lo, hi, 0,1,2,3,4,5,6,7,8,9,10,11,12,13,14,15);
      acc[nt] = __builtin_amdgcn_wmma_f32_16x16x32_bf16(
          false, af, false, bfr, (short)0, acc[nt], false, false);
    }
  }

  int roff = (lane < 16) ? 0 : 8;
#pragma unroll
  for (int nt = 0; nt < 2; ++nt) {
    int col = nt * 16 + lrow;
    if (col < OUT_D) {
      float bb = bias[col];
#pragma unroll
      for (int j = 0; j < 8; ++j) {
        int row = mbase + w * 16 + roff + j;
        out[(size_t)row * OUT_D + col] = acc[nt][j] + bb;
      }
    }
  }
}

// ---------------------------------------------------------------------------
extern "C" void kernel_launch(void* const* d_in, const int* in_sizes, int n_in,
                              void* d_out, int out_size, void* d_ws, size_t ws_size,
                              hipStream_t stream) {
  const float* x      = (const float*)d_in[0];
  const float* coords = (const float*)d_in[1];
  const float* genes  = (const float*)d_in[2];
  const float* W1     = (const float*)d_in[3];
  const float* b1     = (const float*)d_in[4];
  const float* W2     = (const float*)d_in[5];
  const float* b2     = (const float*)d_in[6];
  float* out = (float*)d_out;

  // workspace layout (bytes)
  char* ws = (char*)d_ws;
  float*  mean = (float*)(ws + 0);                         // 128 B
  int*    nidx = (int*)(ws + 256);                         // 512 KB
  __bf16* hid  = (__bf16*)(ws + (1u << 20));               // 16384*3072*2 = 96 MB
  __bf16* W1t  = (__bf16*)(ws + 101711872u);               // 1024*3072*2  = 6 MB
  __bf16* W2t  = (__bf16*)(ws + 108003328u);               // 32*1024*2
  __bf16* hB   = (__bf16*)(ws + 108068864u);               // 16384*1024*2 = 32 MB

  mean_genes<<<1, 256, 0, stream>>>(genes, mean);
  knn_kernel<<<N_CELLS / 256, 256, 0, stream>>>(coords, nidx);
  {
    const size_t total = (size_t)H1 * HID_P + (size_t)OUT_P * H1;
    prep_weights<<<(unsigned)((total + 255) / 256), 256, 0, stream>>>(W1, W2, W1t, W2t);
  }
  pack_hidden<<<(N_CELLS * HID_P) / 256, 256, 0, stream>>>(x, hid);
  covet_kernel<<<N_CELLS / 8, 256, 0, stream>>>(genes, nidx, mean, hid);
  gemm1_relu<<<dim3(H1 / 64, N_CELLS / 128), 256, 0, stream>>>(hid, W1t, b1, hB);
  gemm2_bias<<<N_CELLS / 128, 256, 0, stream>>>(hB, W2t, b2, out);
}
